// VAE_80633716015149
// MI455X (gfx1250) — matrix operand order repaired and checked by reading, not compiled, not run on hardware
//
#include <hip/hip_runtime.h>

// ---------------------------------------------------------------------------
// VAE (LSTM encoder -> reparam -> autoregressive LSTM decoder) for gfx1250.
// B=512, T=512, D=64, L=32, H=128, 4H=512.
// Batch-parallel recurrence: each workgroup owns 32 batch rows, keeps all
// weights + h/c state in LDS, runs all T steps with v_wmma_f32_16x16x32_f16.
// ---------------------------------------------------------------------------

typedef __attribute__((ext_vector_type(16))) _Float16 v16h;
typedef __attribute__((ext_vector_type(8)))  float    v8f;

#define WMMA(a, b, c) \
  __builtin_amdgcn_wmma_f32_16x16x32_f16(false, (a), false, (b), (short)0, (c), false, false)

__device__ __forceinline__ float sigm_f(float x) { return 1.0f / (1.0f + __expf(-x)); }

// A-fragment: 16x32 f16 tile, rows m0..m0+15 (base already offset), ld in elems.
// Layout: lanes 0-15 row M=lane K=0..15; lanes 16-31 K=16..31 (matches load_b_frag).
__device__ __forceinline__ v16h load_a_frag(const _Float16* base, int lda, int lane) {
  int m = lane & 15, hi = lane >> 4;
  v16h a;
#pragma unroll
  for (int e = 0; e < 16; ++e) {
    int k = hi * 16 + e;
    a[e] = base[m * lda + k];
  }
  return a;
}

// B-fragment: 32x16 (KxN) from row-major W[N][K] (i.e. B[k][n] = W[n][k]).
// Layout used here (contiguous k, as load_a_frag): lane holds col n0+(lane&15); lanes 0-15 K=k0..k0+15, lanes 16-31 K=k0+16..k0+31.
__device__ __forceinline__ v16h load_b_frag(const _Float16* W, int ldw, int n0, int k0, int lane) {
  int n = n0 + (lane & 15), hi = lane >> 4;
  const _Float16* p = &W[n * ldw + k0 + hi * 16];
  v16h b;
#pragma unroll
  for (int e = 0; e < 16; ++e) b[e] = p[e];
  return b;
}

__device__ __forceinline__ v8f zero8() {
  v8f z = {0.f, 0.f, 0.f, 0.f, 0.f, 0.f, 0.f, 0.f};
  return z;
}

// ---------------------------------------------------------------------------
// Encoder LSTM: 16 workgroups x 256 threads; each WG owns 32 batch rows.
// ---------------------------------------------------------------------------
__global__ __launch_bounds__(256)
void vae_enc_lstm(const float* __restrict__ x,
                  const float* __restrict__ W_ih, const float* __restrict__ W_hh,
                  const float* __restrict__ b_ih, const float* __restrict__ b_hh,
                  float* __restrict__ hT, int T) {
  constexpr int D = 64, H = 128, G = 512, BM = 32;
  extern __shared__ char smem[];
  _Float16* sWih = (_Float16*)(smem);                       // [512][64]  64 KB
  _Float16* sWhh = (_Float16*)(smem + 65536);               // [512][128] 128 KB
  _Float16* sX   = (_Float16*)(smem + 65536 + 131072);      // [32][64]   4 KB
  _Float16* sH   = (_Float16*)(smem + 65536 + 131072 + 4096);         // [32][128] 8 KB
  float*    sB   = (float*)   (smem + 65536 + 131072 + 4096 + 8192);  // [512]     2 KB
  float*    sC   = (float*)   (smem + 65536 + 131072 + 4096 + 8192 + 2048); // [32][128] 16 KB

  const int tid  = threadIdx.x;
  const int lane = tid & 31;
  const int w    = tid >> 5;     // wave 0..7
  const int b0   = blockIdx.x * BM;
  const int hn   = w;            // this wave's H-column tile (0..7)

  for (int i = tid; i < G * D; i += 256) sWih[i] = (_Float16)W_ih[i];
  for (int i = tid; i < G * H; i += 256) sWhh[i] = (_Float16)W_hh[i];
  for (int i = tid; i < G; i += 256)     sB[i]   = b_ih[i] + b_hh[i];
  for (int i = tid; i < BM * H; i += 256) { sH[i] = (_Float16)0.0f; sC[i] = 0.0f; }

  for (int t = 0; t < T; ++t) {
    __syncthreads();   // prev-step h/c writes (and init) visible
    // stage x[:, t, :] for this batch slice as f16
    for (int i = tid; i < BM * D; i += 256) {
      int r = i >> 6, d = i & 63;
      sX[i] = (_Float16)x[((size_t)(b0 + r) * T + t) * D + d];
    }
    __syncthreads();

    v8f acc[2][4];
#pragma unroll
    for (int mt = 0; mt < 2; ++mt) {
      v16h ax0 = load_a_frag(&sX[(mt * 16) * D + 0],  D, lane);
      v16h ax1 = load_a_frag(&sX[(mt * 16) * D + 32], D, lane);
      v16h ah0 = load_a_frag(&sH[(mt * 16) * H + 0],  H, lane);
      v16h ah1 = load_a_frag(&sH[(mt * 16) * H + 32], H, lane);
      v16h ah2 = load_a_frag(&sH[(mt * 16) * H + 64], H, lane);
      v16h ah3 = load_a_frag(&sH[(mt * 16) * H + 96], H, lane);
#pragma unroll
      for (int g = 0; g < 4; ++g) {
        int n0 = g * H + hn * 16;           // [i|f|g|o] each 128 wide
        v8f c = zero8();
        c = WMMA(ax0, load_b_frag(sWih, D, n0, 0,  lane), c);
        c = WMMA(ax1, load_b_frag(sWih, D, n0, 32, lane), c);
        c = WMMA(ah0, load_b_frag(sWhh, H, n0, 0,  lane), c);
        c = WMMA(ah1, load_b_frag(sWhh, H, n0, 32, lane), c);
        c = WMMA(ah2, load_b_frag(sWhh, H, n0, 64, lane), c);
        c = WMMA(ah3, load_b_frag(sWhh, H, n0, 96, lane), c);
        acc[mt][g] = c;
      }
    }
    __syncthreads();   // all h reads done before we overwrite h

    const int col = hn * 16 + (lane & 15);
    const int hi  = lane >> 4;
#pragma unroll
    for (int mt = 0; mt < 2; ++mt) {
#pragma unroll
      for (int r = 0; r < 8; ++r) {
        int row = mt * 16 + hi * 8 + r;
        float ig = sigm_f(acc[mt][0][r] + sB[col]);
        float fg = sigm_f(acc[mt][1][r] + sB[128 + col]);
        float gg = tanhf (acc[mt][2][r] + sB[256 + col]);
        float og = sigm_f(acc[mt][3][r] + sB[384 + col]);
        float cn = fg * sC[row * H + col] + ig * gg;
        float hv = og * tanhf(cn);
        sC[row * H + col] = cn;
        sH[row * H + col] = (_Float16)hv;
        if (t == T - 1) hT[(size_t)(b0 + row) * H + col] = hv;
      }
    }
  }
}

// ---------------------------------------------------------------------------
// Latent head: mu, log_var, z = mu + eps*exp(0.5*lv), h_dec = z @ init_W^T + b
// ---------------------------------------------------------------------------
__global__ __launch_bounds__(128)
void vae_latent(const float* __restrict__ hT, const float* __restrict__ eps,
                const float* __restrict__ mu_W, const float* __restrict__ mu_b,
                const float* __restrict__ lv_W, const float* __restrict__ lv_b,
                const float* __restrict__ init_W, const float* __restrict__ init_b,
                float* __restrict__ mu_out, float* __restrict__ lv_out,
                float* __restrict__ hdec) {
  constexpr int H = 128, L = 32;
  __shared__ float sh[H];
  __shared__ float sz[L];
  const int b = blockIdx.x, tid = threadIdx.x;
  sh[tid] = hT[(size_t)b * H + tid];
  __syncthreads();
  if (tid < L) {
    float m = mu_b[tid], v = lv_b[tid];
    for (int k = 0; k < H; ++k) {
      m += sh[k] * mu_W[tid * H + k];
      v += sh[k] * lv_W[tid * H + k];
    }
    mu_out[b * L + tid] = m;
    lv_out[b * L + tid] = v;
    sz[tid] = m + eps[b * L + tid] * __expf(0.5f * v);
  }
  __syncthreads();
  float a = init_b[tid];
  for (int l = 0; l < L; ++l) a += sz[l] * init_W[tid * L + l];
  hdec[(size_t)b * H + tid] = a;
}

// ---------------------------------------------------------------------------
// Decoder LSTM + output head, autoregressive.
// ---------------------------------------------------------------------------
__global__ __launch_bounds__(256)
void vae_dec_lstm(const float* __restrict__ hdec,
                  const float* __restrict__ W_ih, const float* __restrict__ W_hh,
                  const float* __restrict__ b_ih, const float* __restrict__ b_hh,
                  const float* __restrict__ out_W, const float* __restrict__ out_b,
                  float* __restrict__ xrec, int T) {
  constexpr int D = 64, H = 128, G = 512, BM = 32;
  extern __shared__ char smem[];
  _Float16* sWih = (_Float16*)(smem);                               // [512][64]
  _Float16* sWhh = (_Float16*)(smem + 65536);                       // [512][128]
  _Float16* sWo  = (_Float16*)(smem + 65536 + 131072);              // [64][128]
  _Float16* sIn  = (_Float16*)(smem + 65536 + 131072 + 16384);      // [32][64]
  _Float16* sH   = (_Float16*)(smem + 65536 + 131072 + 16384 + 4096);        // [32][128]
  float*    sBd  = (float*)   (smem + 65536 + 131072 + 16384 + 4096 + 8192); // [512]
  float*    sOb  = (float*)   (smem + 65536 + 131072 + 16384 + 4096 + 8192 + 2048); // [64]
  float*    sC   = (float*)   (smem + 65536 + 131072 + 16384 + 4096 + 8192 + 2048 + 256); // [32][128]

  const int tid  = threadIdx.x;
  const int lane = tid & 31;
  const int w    = tid >> 5;
  const int b0   = blockIdx.x * BM;
  const int hn   = w;

  for (int i = tid; i < G * D; i += 256) sWih[i] = (_Float16)W_ih[i];
  for (int i = tid; i < G * H; i += 256) sWhh[i] = (_Float16)W_hh[i];
  for (int i = tid; i < D * H; i += 256) sWo[i]  = (_Float16)out_W[i];
  for (int i = tid; i < G; i += 256)     sBd[i]  = b_ih[i] + b_hh[i];
  for (int i = tid; i < D; i += 256)     sOb[i]  = out_b[i];
  for (int i = tid; i < BM * D; i += 256) sIn[i] = (_Float16)0.0f;
  for (int i = tid; i < BM * H; i += 256) {
    sH[i] = (_Float16)hdec[(size_t)b0 * H + i];
    sC[i] = 0.0f;
  }

  for (int t = 0; t < T; ++t) {
    __syncthreads();   // prev-step h / y(inp) writes visible

    v8f acc[2][4];
#pragma unroll
    for (int mt = 0; mt < 2; ++mt) {
      v16h ai0 = load_a_frag(&sIn[(mt * 16) * D + 0],  D, lane);
      v16h ai1 = load_a_frag(&sIn[(mt * 16) * D + 32], D, lane);
      v16h ah0 = load_a_frag(&sH[(mt * 16) * H + 0],   H, lane);
      v16h ah1 = load_a_frag(&sH[(mt * 16) * H + 32],  H, lane);
      v16h ah2 = load_a_frag(&sH[(mt * 16) * H + 64],  H, lane);
      v16h ah3 = load_a_frag(&sH[(mt * 16) * H + 96],  H, lane);
#pragma unroll
      for (int g = 0; g < 4; ++g) {
        int n0 = g * H + hn * 16;
        v8f c = zero8();
        c = WMMA(ai0, load_b_frag(sWih, D, n0, 0,  lane), c);
        c = WMMA(ai1, load_b_frag(sWih, D, n0, 32, lane), c);
        c = WMMA(ah0, load_b_frag(sWhh, H, n0, 0,  lane), c);
        c = WMMA(ah1, load_b_frag(sWhh, H, n0, 32, lane), c);
        c = WMMA(ah2, load_b_frag(sWhh, H, n0, 64, lane), c);
        c = WMMA(ah3, load_b_frag(sWhh, H, n0, 96, lane), c);
        acc[mt][g] = c;
      }
    }
    __syncthreads();   // h reads done

    {
      const int col = hn * 16 + (lane & 15);
      const int hi  = lane >> 4;
#pragma unroll
      for (int mt = 0; mt < 2; ++mt) {
#pragma unroll
        for (int r = 0; r < 8; ++r) {
          int row = mt * 16 + hi * 8 + r;
          float ig = sigm_f(acc[mt][0][r] + sBd[col]);
          float fg = sigm_f(acc[mt][1][r] + sBd[128 + col]);
          float gg = tanhf (acc[mt][2][r] + sBd[256 + col]);
          float og = sigm_f(acc[mt][3][r] + sBd[384 + col]);
          float cn = fg * sC[row * H + col] + ig * gg;
          float hv = og * tanhf(cn);
          sC[row * H + col] = cn;
          sH[row * H + col] = (_Float16)hv;
        }
      }
    }
    __syncthreads();   // new h fully written before y = h @ out_W^T

    {
      const int mt = w >> 2, nd = w & 3;   // 2 M-tiles x 4 D-tiles across 8 waves
      v16h a0 = load_a_frag(&sH[(mt * 16) * H + 0],  H, lane);
      v16h a1 = load_a_frag(&sH[(mt * 16) * H + 32], H, lane);
      v16h a2 = load_a_frag(&sH[(mt * 16) * H + 64], H, lane);
      v16h a3 = load_a_frag(&sH[(mt * 16) * H + 96], H, lane);
      v8f y = zero8();
      y = WMMA(a0, load_b_frag(sWo, H, nd * 16, 0,  lane), y);
      y = WMMA(a1, load_b_frag(sWo, H, nd * 16, 32, lane), y);
      y = WMMA(a2, load_b_frag(sWo, H, nd * 16, 64, lane), y);
      y = WMMA(a3, load_b_frag(sWo, H, nd * 16, 96, lane), y);
      const int colD = nd * 16 + (lane & 15);
      const int hi   = lane >> 4;
#pragma unroll
      for (int r = 0; r < 8; ++r) {
        int row = mt * 16 + hi * 8 + r;
        float yv = y[r] + sOb[colD];
        xrec[((size_t)(b0 + row) * T + t) * D + colD] = yv;
        sIn[row * D + colD] = (_Float16)yv;   // next step's input
      }
    }
  }
}

// ---------------------------------------------------------------------------
extern "C" void kernel_launch(void* const* d_in, const int* in_sizes, int n_in,
                              void* d_out, int out_size, void* d_ws, size_t ws_size,
                              hipStream_t stream) {
  (void)in_sizes; (void)n_in; (void)out_size; (void)ws_size;
  const int B = 512, T = 512, D = 64, L = 32, H = 128;

  const float* x    = (const float*)d_in[0];
  const float* eps  = (const float*)d_in[1];
  const float* eWih = (const float*)d_in[2];
  const float* eWhh = (const float*)d_in[3];
  const float* eBih = (const float*)d_in[4];
  const float* eBhh = (const float*)d_in[5];
  const float* muW  = (const float*)d_in[6];
  const float* muB  = (const float*)d_in[7];
  const float* lvW  = (const float*)d_in[8];
  const float* lvB  = (const float*)d_in[9];
  const float* iW   = (const float*)d_in[10];
  const float* iB   = (const float*)d_in[11];
  const float* dWih = (const float*)d_in[12];
  const float* dWhh = (const float*)d_in[13];
  const float* dBih = (const float*)d_in[14];
  const float* dBhh = (const float*)d_in[15];
  const float* oW   = (const float*)d_in[16];
  const float* oB   = (const float*)d_in[17];

  float* xrec = (float*)d_out;                    // [B,T,D]
  float* muO  = xrec + (size_t)B * T * D;         // [B,L]
  float* lvO  = muO + (size_t)B * L;              // [B,L]

  float* hT   = (float*)d_ws;                     // [B,H]
  float* hdec = hT + (size_t)B * H;               // [B,H]

  const size_t encLds = 65536 + 131072 + 4096 + 8192 + 2048 + 16384;          // 222 KB
  const size_t decLds = 65536 + 131072 + 16384 + 4096 + 8192 + 2048 + 256 + 16384; // ~238 KB

  vae_enc_lstm<<<B / 32, 256, encLds, stream>>>(x, eWih, eWhh, eBih, eBhh, hT, T);
  vae_latent<<<B, 128, 0, stream>>>(hT, eps, muW, muB, lvW, lvB, iW, iB, muO, lvO, hdec);
  vae_dec_lstm<<<B / 32, 256, decLds, stream>>>(hdec, dWih, dWhh, dBih, dBhh, oW, oB, xrec, T);
}
